// DeformableLayer_61074434949228
// MI455X (gfx1250) — matrix operand order repaired and checked by reading, not compiled, not run on hardware
//
#include <hip/hip_runtime.h>
#include <math.h>

typedef __attribute__((ext_vector_type(16))) _Float16 v16h;
typedef __attribute__((ext_vector_type(8)))  float    v8f;

#define WMMA_F16(a, b, c) \
  __builtin_amdgcn_wmma_f32_16x16x32_f16(false, (a), false, (b), (short)0, (c), false, false)

// CDNA5 WMMA 16x16x32 f16 fragment index helpers (ISA 7.12.2, wave32).
// A (16x32, M x K): lane holds row m = lane&15; element (v,h) has
//   k = (v<4?0:16) + hi*8 + (v&3)*2 + h          (hi = lane>>4)
__device__ __forceinline__ int a_kidx(int v, int h, int hi) {
  return ((v & 4) << 2) + hi * 8 + ((v & 3) << 1) + h;
}
// B (32x16, K x N): lane holds col n = lane&15; element (v,h) has
//   k = (v<4?0:16) + hi*8 + (v&3)*2 + h
__device__ __forceinline__ int b_kidx(int v, int h, int hi) {
  return ((v & 4) << 2) + hi * 8 + ((v & 3) << 1) + h;
}
// C/D (16x16 f32): lane holds col n = lane&15; VGPR v holds row m = v + 8*hi.

// ---------------------------------------------------------------------------
// Generic grouped 1x1-conv GEMM:  C[g][o][n] = sum_k W[g%wg][o][k]*X[g][k][n]
// One 16x16 C tile per wave; K stepped by 32.
// ---------------------------------------------------------------------------
__global__ void gemm_gconv_wmma(const float* __restrict__ W, const float* __restrict__ X,
                                float* __restrict__ C, const float* __restrict__ bias,
                                int Gt, int O, int K, int N, int wg) {
  const int lane = threadIdx.x & 31;
  const int nl = lane & 15;
  const int hi = lane >> 4;
  const int tiles_n = N >> 4, tiles_m = O >> 4;
  const long total = (long)Gt * tiles_m * tiles_n;
  long wid = (long)blockIdx.x * (blockDim.x >> 5) + (threadIdx.x >> 5);
  const long wstride = (long)gridDim.x * (blockDim.x >> 5);
  for (long t = wid; t < total; t += wstride) {
    int nt = (int)(t % tiles_n);
    long r = t / tiles_n;
    int mt = (int)(r % tiles_m);
    int g = (int)(r / tiles_m);
    const float* Wg = W + (long)(g % wg) * O * K;
    const float* Xg = X + (long)g * K * N;
    float* Cg = C + (long)g * O * N;
    v8f acc = {};
    for (int kb = 0; kb < K; kb += 32) {
      v16h af, bf;
#pragma unroll
      for (int vv = 0; vv < 8; ++vv) {
#pragma unroll
        for (int h = 0; h < 2; ++h) {
          af[vv * 2 + h] = (_Float16)Wg[(long)(mt * 16 + nl) * K + kb + a_kidx(vv, h, hi)];
          bf[vv * 2 + h] = (_Float16)Xg[(long)(kb + b_kidx(vv, h, hi)) * N + nt * 16 + nl];
        }
      }
      acc = WMMA_F16(af, bf, acc);
    }
#pragma unroll
    for (int vv = 0; vv < 8; ++vv) {
      int m = mt * 16 + vv + 8 * hi;
      float val = acc[vv];
      if (bias) val += bias[m];
      Cg[(long)m * N + nt * 16 + nl] = val;
    }
  }
}

// ---------------------------------------------------------------------------
// Offset net: depthwise 6x6 stride-4 pad-1 conv + exact GELU + 1x1 -> tanh*4
// -> normalized deformed grid (16, 64, 2).  One block per bg.
// ---------------------------------------------------------------------------
__global__ void offset_kernel(const float* __restrict__ q, const float* __restrict__ w1c,
                              const float* __restrict__ b1c, const float* __restrict__ w2c,
                              float* __restrict__ vgrid) {
  __shared__ float act[64 * 64];  // [p][c]
  const int bg = blockIdx.x;
  const int tid = threadIdx.x;
  const float* gq = q + (long)bg * 64 * 1024;
  for (int idx = tid; idx < 4096; idx += 256) {
    int p = idx >> 6, c = idx & 63;
    int oy = p >> 3, ox = p & 7;
    float s = 0.0f;
#pragma unroll
    for (int ky = 0; ky < 6; ++ky) {
      int yy = oy * 4 - 1 + ky;
      if (yy < 0 || yy >= 32) continue;
#pragma unroll
      for (int kx = 0; kx < 6; ++kx) {
        int xx = ox * 4 - 1 + kx;
        if (xx < 0 || xx >= 32) continue;
        s += w1c[c * 36 + ky * 6 + kx] * gq[(long)c * 1024 + yy * 32 + xx];
      }
    }
    s += b1c[c];
    act[p * 64 + c] = 0.5f * s * (1.0f + erff(s * 0.70710678118654752f));  // exact GELU
  }
  __syncthreads();
  if (tid < 128) {
    int p = tid >> 1, comp = tid & 1;
    float s = 0.0f;
#pragma unroll
    for (int c = 0; c < 64; ++c) s += w2c[comp * 64 + c] * act[p * 64 + c];
    float off = tanhf(s) * 4.0f;  // offset_scale = downsample = 4
    int oy = p >> 3, ox = p & 7;
    float coord = (comp == 0 ? (float)ox : (float)oy) + off;
    vgrid[(bg * 64 + p) * 2 + comp] = coord * (2.0f / 7.0f) - 1.0f;  // /(h2-1)
  }
}

// ---------------------------------------------------------------------------
// Bilinear grid-sample with zero padding: kv[b][g*32+c][p]
// ---------------------------------------------------------------------------
__device__ __forceinline__ float samp2d(const float* img, int yy, int xx) {
  if (yy < 0 || yy >= 32 || xx < 0 || xx >= 32) return 0.0f;
  return img[yy * 32 + xx];
}

__global__ void gridsample_kernel(const float* __restrict__ x, const float* __restrict__ vgrid,
                                  float* __restrict__ kv) {
  int idx = blockIdx.x * 256 + threadIdx.x;  // 16*64*32
  if (idx >= 16 * 64 * 32) return;
  int c = idx & 31;
  int p = (idx >> 5) & 63;
  int bg = idx >> 11;
  float g0 = vgrid[(bg * 64 + p) * 2 + 0];
  float g1 = vgrid[(bg * 64 + p) * 2 + 1];
  float fx = (g0 + 1.0f) * 16.0f - 0.5f;  // W/2 = 16
  float fy = (g1 + 1.0f) * 16.0f - 0.5f;
  float x0f = floorf(fx), y0f = floorf(fy);
  float wx1 = fx - x0f, wx0 = 1.0f - wx1;
  float wy1 = fy - y0f, wy0 = 1.0f - wy1;
  int x0 = (int)x0f, y0 = (int)y0f;
  const float* img = x + ((long)bg * 32 + c) * 1024;
  float val = samp2d(img, y0, x0) * wy0 * wx0 + samp2d(img, y0, x0 + 1) * wy0 * wx1 +
              samp2d(img, y0 + 1, x0) * wy1 * wx0 + samp2d(img, y0 + 1, x0 + 1) * wy1 * wx1;
  kv[((long)bg * 32 + c) * 64 + p] = val;
}

// ---------------------------------------------------------------------------
// CPB relative-position-bias MLP: 2 -> 64 -> 64 -> 1 for 16*1024*64 positions.
// Tokens (16 keys of one jt tile) on M; hidden-out 64 on N; K=64 (2 steps).
// Layer-2 on WMMA (dominant GEMM), layers 1/3 on VALU.
// ---------------------------------------------------------------------------
__global__ void cpb_bias_kernel(const float* __restrict__ vgrid, const float* __restrict__ w1,
                                const float* __restrict__ b1, const float* __restrict__ w2,
                                const float* __restrict__ b2, const float* __restrict__ w3,
                                const float* __restrict__ b3, float* __restrict__ bias_out) {
  __shared__ _Float16 sW2[64 * 64];  // (n_out, k_in) row-major
  __shared__ float sw1[128], sb1[64], sb2[64], sw3[64];
  const int tid = threadIdx.x;
  for (int i = tid; i < 4096; i += 256) sW2[i] = (_Float16)w2[i];
  if (tid < 128) sw1[tid] = w1[tid];
  if (tid < 64) { sb1[tid] = b1[tid]; sb2[tid] = b2[tid]; sw3[tid] = w3[tid]; }
  __syncthreads();
  const int bg = blockIdx.x;
  const int iq0 = blockIdx.y * 64;
  const int wave = tid >> 5, lane = tid & 31;
  const int jt = wave & 3, half = wave >> 1 >> 1;  // wave>>2
  const int nl = lane & 15, hi = lane >> 4;
  const float b3v = b3[0];
  // Hoist W2 B-fragments across the query loop
  v16h Bf[4][2];
#pragma unroll
  for (int nt = 0; nt < 4; ++nt)
#pragma unroll
    for (int kb = 0; kb < 2; ++kb)
#pragma unroll
      for (int vv = 0; vv < 8; ++vv)
#pragma unroll
        for (int h = 0; h < 2; ++h)
          Bf[nt][kb][vv * 2 + h] = sW2[(nt * 16 + nl) * 64 + kb * 32 + b_kidx(vv, h, hi)];
  const int jk = jt * 16 + nl;  // this lane's key token (= A-frag row)
  const float gk0 = vgrid[(bg * 64 + jk) * 2 + 0];
  const float gk1 = vgrid[(bg * 64 + jk) * 2 + 1];
  for (int it = 0; it < 32; ++it) {
    const int iq = iq0 + half * 32 + it;
    const int qx = iq & 31, qy = iq >> 5;
    const float p0 = (qx * (2.0f / 31.0f) - 1.0f) - gk0;
    const float p1 = (qy * (2.0f / 31.0f) - 1.0f) - gk1;
    const float c0 = copysignf(log1pf(fabsf(p0)), p0);
    const float c1 = copysignf(log1pf(fabsf(p1)), p1);
    v16h A0, A1;  // layer-1 output straight into A-frag layout
#pragma unroll
    for (int vv = 0; vv < 8; ++vv)
#pragma unroll
      for (int h = 0; h < 2; ++h) {
        int k0 = a_kidx(vv, h, hi);
        A0[vv * 2 + h] = (_Float16)fmaxf(c0 * sw1[k0 * 2] + c1 * sw1[k0 * 2 + 1] + sb1[k0], 0.0f);
        int k1 = 32 + k0;
        A1[vv * 2 + h] = (_Float16)fmaxf(c0 * sw1[k1 * 2] + c1 * sw1[k1 * 2 + 1] + sb1[k1], 0.0f);
      }
    v8f acc0 = {}, acc1 = {}, acc2 = {}, acc3 = {};
    acc0 = WMMA_F16(A0, Bf[0][0], acc0); acc0 = WMMA_F16(A1, Bf[0][1], acc0);
    acc1 = WMMA_F16(A0, Bf[1][0], acc1); acc1 = WMMA_F16(A1, Bf[1][1], acc1);
    acc2 = WMMA_F16(A0, Bf[2][0], acc2); acc2 = WMMA_F16(A1, Bf[2][1], acc2);
    acc3 = WMMA_F16(A0, Bf[3][0], acc3); acc3 = WMMA_F16(A1, Bf[3][1], acc3);
    float part[8];
#pragma unroll
    for (int vv = 0; vv < 8; ++vv) {
      float e;
      part[vv] = 0.0f;
      e = fmaxf(acc0[vv] + sb2[0 * 16 + nl], 0.0f); part[vv] += e * sw3[0 * 16 + nl];
      e = fmaxf(acc1[vv] + sb2[1 * 16 + nl], 0.0f); part[vv] += e * sw3[1 * 16 + nl];
      e = fmaxf(acc2[vv] + sb2[2 * 16 + nl], 0.0f); part[vv] += e * sw3[2 * 16 + nl];
      e = fmaxf(acc3[vv] + sb2[3 * 16 + nl], 0.0f); part[vv] += e * sw3[3 * 16 + nl];
    }
#pragma unroll
    for (int vv = 0; vv < 8; ++vv) {  // reduce over n within each 16-lane half
      float s = part[vv];
      s += __shfl_xor(s, 1, 32);
      s += __shfl_xor(s, 2, 32);
      s += __shfl_xor(s, 4, 32);
      s += __shfl_xor(s, 8, 32);
      part[vv] = s;
    }
    if (nl == 0) {
#pragma unroll
      for (int vv = 0; vv < 8; ++vv)
        bias_out[((long)bg * 1024 + iq) * 64 + jt * 16 + vv + 8 * hi] = part[vv] + b3v;
    }
  }
}

// ---------------------------------------------------------------------------
// Fused attention per (b,h): sim = (q*0.125) K^T + bias; softmax; out = P V.
// Block = 8 waves x 16 queries = 128 queries; K/V staged in LDS as f16.
// ---------------------------------------------------------------------------
__global__ void attn_kernel(const float* __restrict__ q, const float* __restrict__ kmat,
                            const float* __restrict__ vmat, const float* __restrict__ bias,
                            float* __restrict__ out_pre) {
  __shared__ _Float16 sKT[64 * 64];    // [d][jk]  (B-frag source for QK^T)
  __shared__ _Float16 sV[64 * 64];     // [jk][d]  (B-frag source for PV)
  __shared__ _Float16 sP[8][16 * 64];  // per-wave probs [m][jk]
  const int bh = blockIdx.x;           // b*8 + h
  const int qbase = blockIdx.y * 128;
  const int tid = threadIdx.x;
  const float* kh = kmat + (long)bh * 64 * 64;
  const float* vh = vmat + (long)bh * 64 * 64;
  for (int idx = tid; idx < 4096; idx += 256) {
    int d = idx >> 6, jk = idx & 63;
    sKT[idx] = (_Float16)kh[idx];
    sV[jk * 64 + d] = (_Float16)vh[idx];
  }
  __syncthreads();
  const int wave = tid >> 5, lane = tid & 31, nl = lane & 15, hi = lane >> 4;
  const int m0 = qbase + wave * 16;
  const float* qb = q + (long)bh * 64 * 1024;
  v16h Aq[2];
#pragma unroll
  for (int kb = 0; kb < 2; ++kb)
#pragma unroll
    for (int vv = 0; vv < 8; ++vv)
#pragma unroll
      for (int h = 0; h < 2; ++h)
        Aq[kb][vv * 2 + h] =
            (_Float16)(qb[(long)(kb * 32 + a_kidx(vv, h, hi)) * 1024 + m0 + nl] * 0.125f);
  v16h Bk[4][2];
#pragma unroll
  for (int nt = 0; nt < 4; ++nt)
#pragma unroll
    for (int kb = 0; kb < 2; ++kb)
#pragma unroll
      for (int vv = 0; vv < 8; ++vv)
#pragma unroll
        for (int h = 0; h < 2; ++h)
          Bk[nt][kb][vv * 2 + h] = sKT[(kb * 32 + b_kidx(vv, h, hi)) * 64 + nt * 16 + nl];
  v8f S0 = {}, S1 = {}, S2 = {}, S3 = {};
  S0 = WMMA_F16(Aq[0], Bk[0][0], S0); S0 = WMMA_F16(Aq[1], Bk[0][1], S0);
  S1 = WMMA_F16(Aq[0], Bk[1][0], S1); S1 = WMMA_F16(Aq[1], Bk[1][1], S1);
  S2 = WMMA_F16(Aq[0], Bk[2][0], S2); S2 = WMMA_F16(Aq[1], Bk[2][1], S2);
  S3 = WMMA_F16(Aq[0], Bk[3][0], S3); S3 = WMMA_F16(Aq[1], Bk[3][1], S3);
  const float* bb = bias + (long)bh * 1024 * 64;
#pragma unroll
  for (int vv = 0; vv < 8; ++vv) {
    long row = (long)(m0 + vv + 8 * hi) * 64;
    S0[vv] += bb[row + 0 * 16 + nl];
    S1[vv] += bb[row + 1 * 16 + nl];
    S2[vv] += bb[row + 2 * 16 + nl];
    S3[vv] += bb[row + 3 * 16 + nl];
  }
  float rmax[8], rsum[8];
#pragma unroll
  for (int vv = 0; vv < 8; ++vv) {
    float m = fmaxf(fmaxf(S0[vv], S1[vv]), fmaxf(S2[vv], S3[vv]));
    m = fmaxf(m, __shfl_xor(m, 1, 32));
    m = fmaxf(m, __shfl_xor(m, 2, 32));
    m = fmaxf(m, __shfl_xor(m, 4, 32));
    m = fmaxf(m, __shfl_xor(m, 8, 32));
    rmax[vv] = m;
  }
#pragma unroll
  for (int vv = 0; vv < 8; ++vv) {
    S0[vv] = expf(S0[vv] - rmax[vv]);
    S1[vv] = expf(S1[vv] - rmax[vv]);
    S2[vv] = expf(S2[vv] - rmax[vv]);
    S3[vv] = expf(S3[vv] - rmax[vv]);
    float s = S0[vv] + S1[vv] + S2[vv] + S3[vv];
    s += __shfl_xor(s, 1, 32);
    s += __shfl_xor(s, 2, 32);
    s += __shfl_xor(s, 4, 32);
    s += __shfl_xor(s, 8, 32);
    rsum[vv] = 1.0f / s;
  }
#pragma unroll
  for (int vv = 0; vv < 8; ++vv) {  // D layout -> LDS (row-major probs)
    int m = (vv + 8 * hi) * 64;
    sP[wave][m + 0 * 16 + nl] = (_Float16)(S0[vv] * rsum[vv]);
    sP[wave][m + 1 * 16 + nl] = (_Float16)(S1[vv] * rsum[vv]);
    sP[wave][m + 2 * 16 + nl] = (_Float16)(S2[vv] * rsum[vv]);
    sP[wave][m + 3 * 16 + nl] = (_Float16)(S3[vv] * rsum[vv]);
  }
  // wave-local LDS: DS ops are in-order within a wave, no barrier needed
  v16h Ap[2];
#pragma unroll
  for (int kb = 0; kb < 2; ++kb)
#pragma unroll
    for (int vv = 0; vv < 8; ++vv)
#pragma unroll
      for (int h = 0; h < 2; ++h)
        Ap[kb][vv * 2 + h] = sP[wave][nl * 64 + kb * 32 + a_kidx(vv, h, hi)];
  v16h Bv[4][2];
#pragma unroll
  for (int nt = 0; nt < 4; ++nt)
#pragma unroll
    for (int kb = 0; kb < 2; ++kb)
#pragma unroll
      for (int vv = 0; vv < 8; ++vv)
#pragma unroll
        for (int h = 0; h < 2; ++h)
          Bv[nt][kb][vv * 2 + h] = sV[(kb * 32 + b_kidx(vv, h, hi)) * 64 + nt * 16 + nl];
  v8f O0 = {}, O1 = {}, O2 = {}, O3 = {};
  O0 = WMMA_F16(Ap[0], Bv[0][0], O0); O0 = WMMA_F16(Ap[1], Bv[0][1], O0);
  O1 = WMMA_F16(Ap[0], Bv[1][0], O1); O1 = WMMA_F16(Ap[1], Bv[1][1], O1);
  O2 = WMMA_F16(Ap[0], Bv[2][0], O2); O2 = WMMA_F16(Ap[1], Bv[2][1], O2);
  O3 = WMMA_F16(Ap[0], Bv[3][0], O3); O3 = WMMA_F16(Ap[1], Bv[3][1], O3);
#pragma unroll
  for (int vv = 0; vv < 8; ++vv) {
    int iq = m0 + vv + 8 * hi;
    out_pre[((long)bh * 64 + 0 * 16 + nl) * 1024 + iq] = O0[vv];
    out_pre[((long)bh * 64 + 1 * 16 + nl) * 1024 + iq] = O1[vv];
    out_pre[((long)bh * 64 + 2 * 16 + nl) * 1024 + iq] = O2[vv];
    out_pre[((long)bh * 64 + 3 * 16 + nl) * 1024 + iq] = O3[vv];
  }
}

// ---------------------------------------------------------------------------
extern "C" void kernel_launch(void* const* d_in, const int* in_sizes, int n_in,
                              void* d_out, int out_size, void* d_ws, size_t ws_size,
                              hipStream_t stream) {
  (void)in_sizes; (void)n_in; (void)out_size; (void)ws_size;
  const float* x      = (const float*)d_in[0];
  const float* q_w    = (const float*)d_in[1];
  const float* k_w    = (const float*)d_in[2];
  const float* v_w    = (const float*)d_in[3];
  const float* out_w  = (const float*)d_in[4];
  const float* out_b  = (const float*)d_in[5];
  const float* off_w1 = (const float*)d_in[6];
  const float* off_b1 = (const float*)d_in[7];
  const float* off_w2 = (const float*)d_in[8];
  const float* cpb_w1 = (const float*)d_in[9];
  const float* cpb_b1 = (const float*)d_in[10];
  const float* cpb_w2 = (const float*)d_in[11];
  const float* cpb_b2 = (const float*)d_in[12];
  const float* cpb_w3 = (const float*)d_in[13];
  const float* cpb_b3 = (const float*)d_in[14];

  float* ws    = (float*)d_ws;
  float* qbuf  = ws;                   // 2*512*1024 = 1048576
  float* vgrid = qbuf + 1048576;       // 16*64*2    = 2048
  float* kv    = vgrid + 2048;         // 2*256*64   = 32768
  float* kbuf  = kv + 32768;           // 2*512*64   = 65536
  float* vbuf  = kbuf + 65536;         // 65536
  float* biasb = vbuf + 65536;         // 16*1024*64 = 1048576
  float* opre  = biasb + 1048576;      // 2*512*1024 = 1048576

  // q = grouped 1x1 conv: 16 groups, 64x32 @ 32x1024
  gemm_gconv_wmma<<<256, 256, 0, stream>>>(q_w, x, qbuf, nullptr, 16, 64, 32, 1024, 8);
  // offsets + normalized deformed grid
  offset_kernel<<<16, 256, 0, stream>>>(qbuf, off_w1, off_b1, off_w2, vgrid);
  // bilinear sampling of x -> kv
  gridsample_kernel<<<128, 256, 0, stream>>>(x, vgrid, kv);
  // k, v grouped 1x1 convs on kv (N = 64)
  gemm_gconv_wmma<<<32, 256, 0, stream>>>(k_w, kv, kbuf, nullptr, 16, 64, 32, 64, 8);
  gemm_gconv_wmma<<<32, 256, 0, stream>>>(v_w, kv, vbuf, nullptr, 16, 64, 32, 64, 8);
  // CPB relative position bias (dominant GEMM on WMMA)
  cpb_bias_kernel<<<dim3(16, 16), 256, 0, stream>>>(vgrid, cpb_w1, cpb_b1, cpb_w2, cpb_b2,
                                                    cpb_w3, cpb_b3, biasb);
  // fused attention
  attn_kernel<<<dim3(16, 8), 256, 0, stream>>>(qbuf, kbuf, vbuf, biasb, opre);
  // output projection (256x512 @ 512x1024 per batch) + bias
  gemm_gconv_wmma<<<256, 256, 0, stream>>>(out_w, opre, (float*)d_out, out_b, 2, 256, 512, 1024, 1);
}
